// GVAE_19653770346689
// MI455X (gfx1250) — compile-verified, matrix operand order checked by reading, not run on hardware
//
#include <hip/hip_runtime.h>
#include <hip/hip_bf16.h>

// ---------------------------------------------------------------------------
// GVAE graph-GRU, restructured for MI455X (gfx1250, wave32, WMMA):
//   P,Q   = hidden @ [W_top | W_bot]                (WMMA bf16 GEMM, 4096x256x1024)
//   S[b,d]= sum_{s!=d} elu(w[s,d]*(P[b,d]+Q[b,s])+b1)   (LDS-resident Q, ELU pass)
//   agg   = S @ ed_w2 * (1/63) + b2                 (WMMA GEMM, 4096x512x256)
//   G     = agg @ [wr|wi|wh]                        (WMMA GEMM, 4096x256x768)
//   GRU gates + new_hidden                          (elementwise, K=8 matmuls inline)
//   T1,T2 = relu FFN                                (WMMA GEMMs, 4096x256x256)
//   pred  = inputs + T2 @ f3 + f3_b                 (tiny K=256,N=8 reduction)
// A-operands are staged to bf16 once per GEMM so the WMMA inner loop is pure
// b128 loads + v_wmma (no per-iteration conversion VALU).
// ---------------------------------------------------------------------------

typedef __attribute__((ext_vector_type(16))) __bf16 v16bf;
typedef __attribute__((ext_vector_type(8)))  __bf16 v8bf;
typedef __attribute__((ext_vector_type(8)))  float  v8f;

#define BDIM 64
#define NDIM 64
#define HDIM 256
#define FDIM 8
#define ROWS (BDIM * NDIM)   // 4096

__device__ __forceinline__ __bf16 f2bf(float f) { return (__bf16)f; }

// ---- prep: dense edge-weight table w2[s][d] = max(edges[e],axis=-1), 0 on diag
__global__ void build_w2_kernel(const float* __restrict__ edges, float* __restrict__ w2) {
  int idx = blockIdx.x * blockDim.x + threadIdx.x;     // 4096
  if (idx >= NDIM * NDIM) return;
  int s = idx >> 6, d = idx & 63;
  float v = 0.0f;
  if (s != d) {
    int e = s * (NDIM - 1) + d - (d > s ? 1 : 0);
    v = fmaxf(edges[2 * e + 0], edges[2 * e + 1]);
  }
  w2[idx] = v;
}

// ---- prep: dst[n*K + k] = bf16(src[k*srcStride + n])   (N-major bf16 weights)
__global__ void transpose_to_bf16(const float* __restrict__ src,
                                  __bf16* __restrict__ dst,
                                  int K, int Ncols, int srcStride) {
  int idx = blockIdx.x * blockDim.x + threadIdx.x;
  if (idx >= K * Ncols) return;
  int n = idx / K, k = idx - n * K;
  dst[idx] = f2bf(src[(size_t)k * srcStride + n]);
}

// ---- stage an f32 activation matrix to bf16 (elementwise, vectorized x4)
__global__ void cvt_f32_bf16(const float4* __restrict__ src, __bf16* __restrict__ dst,
                             int n4) {
  int i = blockIdx.x * blockDim.x + threadIdx.x;
  if (i >= n4) return;
  float4 v = src[i];
  __bf16* d = dst + 4 * (size_t)i;
  d[0] = f2bf(v.x); d[1] = f2bf(v.y); d[2] = f2bf(v.z); d[3] = f2bf(v.w);
}

// ---- WMMA bf16 GEMM: C(MxN,f32) = A(MxK,bf16) @ Bt^T (Bt is N-major bf16).
// One wave computes a 16x32 slab (two 16x16 WMMA tiles sharing the A fragment).
// EPI: 0 = none, 1 = y*scale + bias[n], 2 = relu(y + bias[n])
template<int EPI>
__global__ void wmma_gemm_bf16(const __bf16* __restrict__ A,
                               const __bf16* __restrict__ Bt,
                               float* __restrict__ C,
                               const float* __restrict__ bias, float scale,
                               int M, int K, int Ncols, int tilesTotal) {
  const int lane   = threadIdx.x & 31;
  const int tileId = blockIdx.x * blockDim.y + threadIdx.y;   // wave-uniform
  if (tileId >= tilesTotal) return;
  const int tilesN2 = Ncols >> 5;                             // 32-wide N slabs
  const int tm = tileId / tilesN2;
  const int tn = tileId - tm * tilesN2;
  const int mrow = (tm << 4) + (lane & 15);
  const int n0   = (tn << 5) + (lane & 15);                   // first tile's column
  const int kb   = (lane >> 4) << 3;          // K half-offset per ISA 16-bit A layout

  const __bf16* arow  = A  + (size_t)mrow * K + kb;
  const __bf16* b0row = Bt + (size_t)n0 * K + kb;
  const __bf16* b1row = b0row + (size_t)16 * K;

  v8f acc0 = {}, acc1 = {};
  for (int k0 = 0; k0 < K; k0 += 32) {
    v8bf alo = *(const v8bf*)(arow + k0);
    v8bf ahi = *(const v8bf*)(arow + k0 + 16);
    v8bf b0l = *(const v8bf*)(b0row + k0);
    v8bf b0h = *(const v8bf*)(b0row + k0 + 16);
    v8bf b1l = *(const v8bf*)(b1row + k0);
    v8bf b1h = *(const v8bf*)(b1row + k0 + 16);
    v16bf a, b0, b1;
#pragma unroll
    for (int i = 0; i < 8; ++i) {
      a[i]  = alo[i]; a[i + 8]  = ahi[i];
      b0[i] = b0l[i]; b0[i + 8] = b0h[i];
      b1[i] = b1l[i]; b1[i + 8] = b1h[i];
    }
    acc0 = __builtin_amdgcn_wmma_f32_16x16x32_bf16(false, a, false, b0,
                                                   (short)0, acc0, false, false);
    acc1 = __builtin_amdgcn_wmma_f32_16x16x32_bf16(false, a, false, b1,
                                                   (short)0, acc1, false, false);
  }

  const int rbase = (tm << 4) + ((lane >> 4) << 3);
  const int c0 = n0, c1 = n0 + 16;
#pragma unroll
  for (int v = 0; v < 8; ++v) {
    float y0 = acc0[v], y1 = acc1[v];
    if (EPI == 1) { y0 = y0 * scale + bias[c0]; y1 = y1 * scale + bias[c1]; }
    if (EPI == 2) { y0 = fmaxf(y0 + bias[c0], 0.0f); y1 = fmaxf(y1 + bias[c1], 0.0f); }
    float* crow = C + (size_t)(rbase + v) * Ncols;
    crow[c0] = y0;
    crow[c1] = y1;
  }
}

// ---- edge pass: one workgroup per batch b; Q[b] (128KB) + w2 (16KB) in LDS.
// thread t owns channel t (0..511). S[b,d,t] = sum_{s!=d} elu(w2[s,d]*(P+Q)+b1)
__global__ void edge_aggregate(const float* __restrict__ PQ,   // 4096 x 1024 (P | Q)
                               const float* __restrict__ w2,   // 64 x 64
                               const float* __restrict__ b1,   // 512
                               float* __restrict__ S) {        // 4096 x 512
  extern __shared__ float smem[];
  float* Qs = smem;                       // 64*512
  float* Ws = smem + NDIM * 512;          // 64*64
  const int b = blockIdx.x;
  const int t = threadIdx.x;              // 0..511

  for (int i = t; i < NDIM * 512; i += 512) {
    int nn = i >> 9, c = i & 511;
    Qs[i] = PQ[((size_t)(b * NDIM + nn)) * 1024 + 512 + c];
  }
  for (int i = t; i < NDIM * NDIM; i += 512) Ws[i] = w2[i];
  const float b1c = b1[t];
  __syncthreads();

  for (int d = 0; d < NDIM; ++d) {
    __builtin_prefetch(&PQ[((size_t)(b * NDIM + ((d + 1) & 63))) * 1024 + t], 0, 0);
    const float p = PQ[((size_t)(b * NDIM + d)) * 1024 + t];
    float acc = 0.0f;
    for (int s = 0; s < NDIM; ++s) {
      if (s == d) continue;               // wave-uniform branch
      float x = Ws[s * NDIM + d] * (p + Qs[s * 512 + t]) + b1c;
      acc += (x > 0.0f) ? x : (__expf(x) - 1.0f);
    }
    S[((size_t)(b * NDIM + d)) * 512 + t] = acc;
  }
}

// ---- GRU gate fusion; K=8 input projections done inline per output channel
__global__ void gru_kernel(const float* __restrict__ inputs,   // 4096 x 8
                           const float* __restrict__ hidden,   // 4096 x 256
                           const float* __restrict__ G,        // 4096 x 768 [r|i|n]
                           const float* __restrict__ ir_w, const float* __restrict__ ir_b,
                           const float* __restrict__ ii_w, const float* __restrict__ ii_b,
                           const float* __restrict__ in_w, const float* __restrict__ in_b,
                           float* __restrict__ outNH,          // d_out slice
                           float* __restrict__ NH) {           // ws copy for FFN
  const int row = blockIdx.x;             // 0..4095
  const int c   = threadIdx.x;            // 0..255
  float in8[FDIM];
#pragma unroll
  for (int k = 0; k < FDIM; ++k) in8[k] = inputs[row * FDIM + k];
  float ur = ir_b[c], ui = ii_b[c], un = in_b[c];
#pragma unroll
  for (int k = 0; k < FDIM; ++k) {
    ur += in8[k] * ir_w[k * HDIM + c];
    ui += in8[k] * ii_w[k * HDIM + c];
    un += in8[k] * in_w[k * HDIM + c];
  }
  const float* g = G + (size_t)row * 768;
  float r  = 1.0f / (1.0f + __expf(-(ur + g[c])));
  float ig = 1.0f / (1.0f + __expf(-(ui + g[HDIM + c])));
  float nn = tanhf(un + r * g[2 * HDIM + c]);
  float nh = (1.0f - ig) * nn + ig * hidden[(size_t)row * HDIM + c];
  outNH[(size_t)row * HDIM + c] = nh;
  NH[(size_t)row * HDIM + c]    = nh;
}

// ---- final projection: pred = inputs + T2 @ f3 + f3_b   (K=256, N=8)
__global__ void final_proj(const float* __restrict__ inputs,
                           const float* __restrict__ T2,      // 4096 x 256
                           const float* __restrict__ f3,      // 256 x 8
                           const float* __restrict__ f3_b,
                           float* __restrict__ pred) {        // 4096 x 8
  int idx = blockIdx.x * blockDim.x + threadIdx.x;
  if (idx >= ROWS * FDIM) return;
  int row = idx >> 3, f = idx & 7;
  float acc = f3_b[f];
  const float* t = T2 + (size_t)row * HDIM;
  for (int k = 0; k < HDIM; ++k) acc += t[k] * f3[k * FDIM + f];
  pred[idx] = inputs[idx] + acc;
}

extern "C" void kernel_launch(void* const* d_in, const int* in_sizes, int n_in,
                              void* d_out, int out_size, void* d_ws, size_t ws_size,
                              hipStream_t stream) {
  (void)in_sizes; (void)n_in; (void)out_size; (void)ws_size;
  const float* inputs = (const float*)d_in[0];
  const float* hidden = (const float*)d_in[1];
  const float* edges  = (const float*)d_in[2];
  // d_in[3] = edge_index: unused (edge enumeration recomputed analytically)
  const float* ed_w1  = (const float*)d_in[4];
  const float* ed_b1  = (const float*)d_in[5];
  const float* ed_w2  = (const float*)d_in[6];
  const float* ed_b2  = (const float*)d_in[7];
  const float* wr     = (const float*)d_in[8];
  const float* wi     = (const float*)d_in[9];
  const float* wh     = (const float*)d_in[10];
  const float* ir_w   = (const float*)d_in[11];
  const float* ir_b   = (const float*)d_in[12];
  const float* ii_w   = (const float*)d_in[13];
  const float* ii_b   = (const float*)d_in[14];
  const float* in_w   = (const float*)d_in[15];
  const float* in_b   = (const float*)d_in[16];
  const float* f1_w   = (const float*)d_in[17];
  const float* f1_b   = (const float*)d_in[18];
  const float* f2_w   = (const float*)d_in[19];
  const float* f2_b   = (const float*)d_in[20];
  const float* f3_w   = (const float*)d_in[21];
  const float* f3_b   = (const float*)d_in[22];

  // ---- workspace carve-out (peak ~34 MB, regions reused across stages)
  char* ws = (char*)d_ws;
  size_t o = 0;
  auto take = [&](size_t bytes) -> char* {
    o = (o + 255) & ~size_t(255);
    char* p = ws + o; o += bytes; return p;
  };
  __bf16* W1t = (__bf16*)take((size_t)1024 * 256 * 2);
  __bf16* W2t = (__bf16*)take((size_t)256 * 512 * 2);
  __bf16* Wgt = (__bf16*)take((size_t)768 * 256 * 2);
  __bf16* F1t = (__bf16*)take((size_t)256 * 256 * 2);
  __bf16* F2t = (__bf16*)take((size_t)256 * 256 * 2);
  float* w2buf = (float*)take((size_t)NDIM * NDIM * 4);
  float* regA  = (float*)take((size_t)ROWS * 1024 * 4);  // PQ, later G
  float* regB  = (float*)take((size_t)ROWS * 512 * 4);   // S, later NH|T1
  float* regC  = (float*)take((size_t)ROWS * 256 * 4);   // agg, later T2
  __bf16* Abf  = (__bf16*)take((size_t)ROWS * 512 * 2);  // staged GEMM A operand
  float* PQ  = regA;
  float* G   = regA;                      // PQ dead before G is written
  float* S   = regB;
  float* NH  = regB;                      // S dead before NH is written
  float* T1  = regB + (size_t)ROWS * HDIM;
  float* agg = regC;
  float* T2  = regC;                      // agg dead before T2 is written

  float* outPred = (float*)d_out;
  float* outNH   = (float*)d_out + (size_t)ROWS * FDIM;

  // ---- prep: edge-weight table + bf16 N-major weights
  build_w2_kernel<<<(NDIM * NDIM + 255) / 256, 256, 0, stream>>>(edges, w2buf);
  auto tgrid = [](int elems) { return dim3((elems + 255) / 256); };
  // W1t rows 0..511 <- W_top = ed_w1[0:256,:], rows 512..1023 <- W_bot = ed_w1[256:,:]
  transpose_to_bf16<<<tgrid(256 * 512), 256, 0, stream>>>(ed_w1,             W1t,             256, 512, 512);
  transpose_to_bf16<<<tgrid(256 * 512), 256, 0, stream>>>(ed_w1 + 256 * 512, W1t + 512 * 256, 256, 512, 512);
  transpose_to_bf16<<<tgrid(512 * 256), 256, 0, stream>>>(ed_w2, W2t, 512, 256, 256);
  transpose_to_bf16<<<tgrid(256 * 256), 256, 0, stream>>>(wr, Wgt,             256, 256, 256);
  transpose_to_bf16<<<tgrid(256 * 256), 256, 0, stream>>>(wi, Wgt + 256 * 256, 256, 256, 256);
  transpose_to_bf16<<<tgrid(256 * 256), 256, 0, stream>>>(wh, Wgt + 512 * 256, 256, 256, 256);
  transpose_to_bf16<<<tgrid(256 * 256), 256, 0, stream>>>(f1_w, F1t, 256, 256, 256);
  transpose_to_bf16<<<tgrid(256 * 256), 256, 0, stream>>>(f2_w, F2t, 256, 256, 256);

  dim3 gemmBlk(32, 8);                    // 8 waves, one 16x32 slab each
  auto ggrid = [](int tiles) { return dim3((tiles + 7) / 8); };
  auto cgrid = [](int n4) { return dim3((n4 + 255) / 256); };

  // ---- PQ = hidden @ [W_top | W_bot]   (M=4096, K=256, N=1024)
  cvt_f32_bf16<<<cgrid(ROWS * 64), 256, 0, stream>>>((const float4*)hidden, Abf, ROWS * 64);
  int tB = (ROWS / 16) * (1024 / 32);
  wmma_gemm_bf16<0><<<ggrid(tB), gemmBlk, 0, stream>>>(Abf, W1t, PQ, nullptr, 0.f,
                                                       ROWS, 256, 1024, tB);

  // ---- edge ELU + destination-sum, Q[b] resident in LDS (144 KB/WG < 320 KB)
  size_t smemBytes = ((size_t)NDIM * 512 + (size_t)NDIM * NDIM) * sizeof(float);
  (void)hipFuncSetAttribute((const void*)edge_aggregate,
                            hipFuncAttributeMaxDynamicSharedMemorySize, (int)smemBytes);
  edge_aggregate<<<BDIM, 512, smemBytes, stream>>>(PQ, w2buf, ed_b1, S);

  // ---- agg = S @ ed_w2 / 63 + b2       (M=4096, K=512, N=256)
  cvt_f32_bf16<<<cgrid(ROWS * 128), 256, 0, stream>>>((const float4*)S, Abf, ROWS * 128);
  int tD = (ROWS / 16) * (HDIM / 32);
  wmma_gemm_bf16<1><<<ggrid(tD), gemmBlk, 0, stream>>>(Abf, W2t, agg, ed_b2, 1.0f / 63.0f,
                                                       ROWS, 512, HDIM, tD);

  // ---- G = agg @ [wr|wi|wh]            (M=4096, K=256, N=768)
  cvt_f32_bf16<<<cgrid(ROWS * 64), 256, 0, stream>>>((const float4*)agg, Abf, ROWS * 64);
  int tE = (ROWS / 16) * (768 / 32);
  wmma_gemm_bf16<0><<<ggrid(tE), gemmBlk, 0, stream>>>(Abf, Wgt, G, nullptr, 0.f,
                                                       ROWS, 256, 768, tE);

  // ---- GRU gates -> new_hidden (to d_out and ws)
  gru_kernel<<<ROWS, HDIM, 0, stream>>>(inputs, hidden, G, ir_w, ir_b, ii_w, ii_b,
                                        in_w, in_b, outNH, NH);

  // ---- FFN: T1 = relu(NH@f1+b), T2 = relu(T1@f2+b)
  int tF = (ROWS / 16) * (HDIM / 32);
  cvt_f32_bf16<<<cgrid(ROWS * 64), 256, 0, stream>>>((const float4*)NH, Abf, ROWS * 64);
  wmma_gemm_bf16<2><<<ggrid(tF), gemmBlk, 0, stream>>>(Abf, F1t, T1, f1_b, 0.f,
                                                       ROWS, HDIM, HDIM, tF);
  cvt_f32_bf16<<<cgrid(ROWS * 64), 256, 0, stream>>>((const float4*)T1, Abf, ROWS * 64);
  wmma_gemm_bf16<2><<<ggrid(tF), gemmBlk, 0, stream>>>(Abf, F2t, T2, f2_b, 0.f,
                                                       ROWS, HDIM, HDIM, tF);

  // ---- pred = inputs + T2 @ f3 + f3_b
  final_proj<<<(ROWS * FDIM + 255) / 256, 256, 0, stream>>>(inputs, T2, f3_w, f3_b, outPred);
}
